// TechBookGAT_18674517803653
// MI455X (gfx1250) — compile-verified, matrix operand order checked by reading, not run on hardware
//
#include <hip/hip_runtime.h>

typedef __attribute__((ext_vector_type(16))) __bf16 v16bf;
typedef __attribute__((ext_vector_type(8)))  float  v8f;
typedef __attribute__((ext_vector_type(4)))  int    v4i;

#define NODES 20000
#define EDGES 320000
#define EETOT (EDGES + NODES)
#define INDIM 1536
#define HIDC  512
#define MIDC  128
#define OUTC  5
#define HEADS 8
#define NEG_SLOPE 0.2f

// GEMM macro-tile config
#define BM 128
#define BN 64
#define BK 32
#define LDS_STRIDE 40   // 32 + 8 pad (80B rows -> conflict-free b128 reads)

#ifndef __has_builtin
#define __has_builtin(x) 0
#endif
#if __has_builtin(__builtin_amdgcn_global_load_async_to_lds_b128) && \
    __has_builtin(__builtin_amdgcn_s_wait_asynccnt)
#define USE_ASYNC_LDS 1
#else
#define USE_ASYNC_LDS 0
#endif

static __device__ __forceinline__ unsigned short f2bf(float f) {
    unsigned u = __float_as_uint(f);
    u += 0x7FFFu + ((u >> 16) & 1u);          // round-to-nearest-even
    return (unsigned short)(u >> 16);
}
// monotonic float<->uint mapping so unsigned atomicMax == float max
static __device__ __forceinline__ unsigned keyenc(float f) {
    unsigned u = __float_as_uint(f);
    return (u & 0x80000000u) ? ~u : (u | 0x80000000u);
}
static __device__ __forceinline__ float keydec(unsigned k) {
    unsigned u = (k & 0x80000000u) ? (k ^ 0x80000000u) : ~k;
    return __uint_as_float(u);
}
static __device__ __forceinline__ void edge_sd(const int* __restrict__ ei, long e, int& s, int& d) {
    if (e < EDGES) { s = ei[e]; d = ei[EDGES + e]; }
    else           { s = d = (int)(e - EDGES); }   // appended self-loops
}

// ---------------- conversion / fill kernels ----------------
// vectorized: 8 elements per thread (n must be a multiple of 8 -- true for all uses)
__global__ void gat_f32_to_bf16_v8(const float4* __restrict__ in, uint4* __restrict__ out, long n8) {
    long i = (long)blockIdx.x * blockDim.x + threadIdx.x;
    if (i >= n8) return;
    float4 f0 = in[2 * i], f1 = in[2 * i + 1];
    union { unsigned short s[8]; uint4 q; } r;
    r.s[0] = f2bf(f0.x); r.s[1] = f2bf(f0.y); r.s[2] = f2bf(f0.z); r.s[3] = f2bf(f0.w);
    r.s[4] = f2bf(f1.x); r.s[5] = f2bf(f1.y); r.s[6] = f2bf(f1.z); r.s[7] = f2bf(f1.w);
    out[i] = r.q;
}
// in: [rows, cols] row-major fp32 -> out: [cols, rows] row-major bf16 (B transposed)
__global__ void gat_f32_to_bf16_T(const float* __restrict__ in, unsigned short* __restrict__ out,
                                  int rows, int cols) {
    long i = (long)blockIdx.x * blockDim.x + threadIdx.x;
    if (i >= (long)rows * cols) return;
    int r = (int)(i / cols), c = (int)(i % cols);
    out[(long)c * rows + r] = f2bf(in[i]);
}
__global__ void gat_zero_f32(float* __restrict__ p, long n) {
    long i = (long)blockIdx.x * blockDim.x + threadIdx.x;
    if (i < n) p[i] = 0.0f;
}
__global__ void gat_zero_u32(unsigned* __restrict__ p, long n) {
    long i = (long)blockIdx.x * blockDim.x + threadIdx.x;
    if (i < n) p[i] = 0u;   // keyenc of any finite float > 0, so 0 == -inf sentinel
}

#if USE_ASYNC_LDS
static __device__ __forceinline__ void async_cp16(const unsigned short* g, unsigned short* l) {
    __builtin_amdgcn_global_load_async_to_lds_b128(
        (__attribute__((address_space(1))) v4i*)(unsigned long long)(const void*)g,
        (__attribute__((address_space(3))) v4i*)l,
        0, 0);
}
#endif

// -------- LDS double-buffered bf16 WMMA GEMM: C[M,N] = A[M,K] * Bt[N,K]^T --------
// Block: 256 threads = 8 waves (4x2). Macro-tile BMxBN, wave-tile 32x32 (4 accum).
// Requires N%BN==0, K%BK==0; M handled with clamped loads + guarded stores.
__global__ void __launch_bounds__(256) gat_wmma_gemm_tiled(
        const unsigned short* __restrict__ A,    // [M,K] bf16
        const unsigned short* __restrict__ Bt,   // [N,K] bf16 (weights pre-transposed)
        float* __restrict__ C, int M, int N, int K) {
    __shared__ __align__(16) unsigned short As[2][BM * LDS_STRIDE];
    __shared__ __align__(16) unsigned short Bs[2][BN * LDS_STRIDE];

    const int tid  = threadIdx.x;
    const int lane = tid & 31;
    const int wave = tid >> 5;
    const int wm   = wave >> 1;          // 0..3  (M sub-block of 32)
    const int wn   = wave & 1;           // 0..1  (N sub-block of 32)
    const int half = lane >> 4;          // K-half selector for fragments
    const int idx  = lane & 15;          // A row / B col within 16-tile
    const int gm0  = blockIdx.y * BM;
    const int gn0  = blockIdx.x * BN;

    // ---- staging assignment: 16B chunk per slot ----
    const int srow = tid >> 2;           // 0..63
    const int sq   = (tid & 3) * 8;      // element offset of 16B chunk within BK row
    const int ar0  = min(gm0 + srow,      M - 1);   // clamped (EXEC stays uniform)
    const int ar1  = min(gm0 + srow + 64, M - 1);
    const int br   = gn0 + srow;
    const unsigned short* Ap0 = A  + (size_t)ar0 * K + sq;
    const unsigned short* Ap1 = A  + (size_t)ar1 * K + sq;
    const unsigned short* Bp  = Bt + (size_t)br  * K + sq;
    unsigned short* LA0[2] = { &As[0][(srow)      * LDS_STRIDE + sq], &As[1][(srow)      * LDS_STRIDE + sq] };
    unsigned short* LA1[2] = { &As[0][(srow + 64) * LDS_STRIDE + sq], &As[1][(srow + 64) * LDS_STRIDE + sq] };
    unsigned short* LB [2] = { &Bs[0][(srow)      * LDS_STRIDE + sq], &Bs[1][(srow)      * LDS_STRIDE + sq] };

    v8f acc[4];
#pragma unroll
    for (int i = 0; i < 4; ++i) acc[i] = (v8f){0.f,0.f,0.f,0.f,0.f,0.f,0.f,0.f};

    const int KT = K / BK;

    // prologue: stage K-step 0 into buffer 0
#if USE_ASYNC_LDS
    async_cp16(Ap0, LA0[0]);
    async_cp16(Ap1, LA1[0]);
    async_cp16(Bp,  LB [0]);
    __builtin_amdgcn_s_wait_asynccnt(0);
#else
    uint4 ra0, ra1, rb;
    ra0 = *(const uint4*)(Ap0);
    ra1 = *(const uint4*)(Ap1);
    rb  = *(const uint4*)(Bp);
    *(uint4*)(LA0[0]) = ra0;
    *(uint4*)(LA1[0]) = ra1;
    *(uint4*)(LB [0]) = rb;
#endif
    __syncthreads();

    for (int kt = 0; kt < KT; ++kt) {
        const int cur = kt & 1;
        const int nxt = cur ^ 1;
        const bool more = (kt + 1) < KT;
        if (more) {                       // issue next K-step early (hide under WMMAs)
            const int ko = (kt + 1) * BK;
#if USE_ASYNC_LDS
            async_cp16(Ap0 + ko, LA0[nxt]);
            async_cp16(Ap1 + ko, LA1[nxt]);
            async_cp16(Bp  + ko, LB [nxt]);
#else
            ra0 = *(const uint4*)(Ap0 + ko);
            ra1 = *(const uint4*)(Ap1 + ko);
            rb  = *(const uint4*)(Bp  + ko);
#endif
        }
        // fragment loads from LDS + 4 WMMAs (2x2 reuse)
        union Frag { v16bf v; uint4 q[2]; } fa[2], fb[2];
#pragma unroll
        for (int s = 0; s < 2; ++s) {
            const int r  = (wm * 32 + s * 16 + idx) * LDS_STRIDE;
            fa[s].q[0] = *(const uint4*)(&As[cur][r + half * 8]);
            fa[s].q[1] = *(const uint4*)(&As[cur][r + 16 + half * 8]);
            const int rn = (wn * 32 + s * 16 + idx) * LDS_STRIDE;
            fb[s].q[0] = *(const uint4*)(&Bs[cur][rn + half * 8]);
            fb[s].q[1] = *(const uint4*)(&Bs[cur][rn + 16 + half * 8]);
        }
        acc[0] = __builtin_amdgcn_wmma_f32_16x16x32_bf16(false, fa[0].v, false, fb[0].v, (short)0, acc[0], false, false);
        acc[1] = __builtin_amdgcn_wmma_f32_16x16x32_bf16(false, fa[0].v, false, fb[1].v, (short)0, acc[1], false, false);
        acc[2] = __builtin_amdgcn_wmma_f32_16x16x32_bf16(false, fa[1].v, false, fb[0].v, (short)0, acc[2], false, false);
        acc[3] = __builtin_amdgcn_wmma_f32_16x16x32_bf16(false, fa[1].v, false, fb[1].v, (short)0, acc[3], false, false);

        if (more) {
#if USE_ASYNC_LDS
            __builtin_amdgcn_s_wait_asynccnt(0);   // LDS writes done before barrier publishes them
#else
            *(uint4*)(LA0[nxt]) = ra0;
            *(uint4*)(LA1[nxt]) = ra1;
            *(uint4*)(LB [nxt]) = rb;
#endif
        }
        __syncthreads();
    }

    // epilogue: C/D layout -> lanes 0-15 rows 0..7, lanes 16-31 rows 8..15 per VGPR
    const bool full = (gm0 + BM) <= M;   // block-uniform fast path (156 of 157 blocks)
#pragma unroll
    for (int s = 0; s < 2; ++s) {
#pragma unroll
        for (int t = 0; t < 2; ++t) {
            const int rbase = gm0 + wm * 32 + s * 16 + half * 8;
            const int col   = gn0 + wn * 32 + t * 16 + idx;
            const v8f a = acc[s * 2 + t];
            if (full) {
#pragma unroll
                for (int v = 0; v < 8; ++v)
                    C[(size_t)(rbase + v) * N + col] = a[v];
            } else {
#pragma unroll
                for (int v = 0; v < 8; ++v) {
                    const int gr = rbase + v;
                    if (gr < M) C[(size_t)gr * N + col] = a[v];
                }
            }
        }
    }
}

// ---------------- attention logits per node/head ----------------
__global__ void gat_node_logits(const float* __restrict__ h, const float* __restrict__ asrc,
                                const float* __restrict__ adst, float* __restrict__ als,
                                float* __restrict__ ald, int n, int H, int Cc) {
    long t = (long)blockIdx.x * blockDim.x + threadIdx.x;
    if (t >= (long)n * H) return;
    int node = (int)(t / H), hd = (int)(t % H);
    const float* hp  = h + (size_t)node * H * Cc + (size_t)hd * Cc;
    const float* asp = asrc + (size_t)hd * Cc;
    const float* adp = adst + (size_t)hd * Cc;
    float ss = 0.f, sd = 0.f;
    for (int c = 0; c < Cc; ++c) { float v = hp[c]; ss += v * asp[c]; sd += v * adp[c]; }
    als[t] = ss; ald[t] = sd;
}

// ---------------- segment softmax over incoming edges ----------------
__global__ void gat_edge_max(const int* __restrict__ ei, const float* __restrict__ als,
                             const float* __restrict__ ald, unsigned* __restrict__ emax, int H) {
    long t = (long)blockIdx.x * blockDim.x + threadIdx.x;
    if (t >= (long)EETOT * H) return;
    long e = t / H; int hd = (int)(t % H); int s, d; edge_sd(ei, e, s, d);
    float v = als[(long)s * H + hd] + ald[(long)d * H + hd];
    v = v >= 0.f ? v : NEG_SLOPE * v;
    atomicMax(&emax[(long)d * H + hd], keyenc(v));
}
__global__ void gat_edge_expsum(const int* __restrict__ ei, const float* __restrict__ als,
                                const float* __restrict__ ald, const unsigned* __restrict__ emax,
                                float* __restrict__ den, int H) {
    long t = (long)blockIdx.x * blockDim.x + threadIdx.x;
    if (t >= (long)EETOT * H) return;
    long e = t / H; int hd = (int)(t % H); int s, d; edge_sd(ei, e, s, d);
    float v = als[(long)s * H + hd] + ald[(long)d * H + hd];
    v = v >= 0.f ? v : NEG_SLOPE * v;
    float ex = __expf(v - keydec(emax[(long)d * H + hd]));
    atomicAdd(&den[(long)d * H + hd], ex);
}
__global__ void gat_edge_aggregate(const int* __restrict__ ei, const float* __restrict__ als,
                                   const float* __restrict__ ald, const unsigned* __restrict__ emax,
                                   const float* __restrict__ den, const float* __restrict__ h,
                                   float* __restrict__ agg, int H, int Cc) {
    long t = (long)blockIdx.x * blockDim.x + threadIdx.x;
    if (t >= (long)EETOT * H) return;
    long e = t / H; int hd = (int)(t % H); int s, d; edge_sd(ei, e, s, d);
    float v = als[(long)s * H + hd] + ald[(long)d * H + hd];
    v = v >= 0.f ? v : NEG_SLOPE * v;
    float alpha = __expf(v - keydec(emax[(long)d * H + hd])) /
                  (den[(long)d * H + hd] + 1e-16f);
    const float* hp = h   + (size_t)s * H * Cc + (size_t)hd * Cc;
    float*       op = agg + (size_t)d * H * Cc + (size_t)hd * Cc;
    for (int c = 0; c < Cc; ++c) atomicAdd(&op[c], hp[c] * alpha);
}

// ---------------- fused bias + relu + batchnorm (+ bf16 copy for next GEMM) ----------------
__global__ void gat_bias_relu_bn(const float* __restrict__ agg, const float* __restrict__ bias,
                                 const float* __restrict__ g, const float* __restrict__ be,
                                 const float* __restrict__ mean, const float* __restrict__ var,
                                 float* __restrict__ actf, unsigned short* __restrict__ actb,
                                 long n, int ch) {
    long i = (long)blockIdx.x * blockDim.x + threadIdx.x;
    if (i >= n) return;
    int c = (int)(i % ch);
    float v = agg[i] + bias[c];
    v = v > 0.f ? v : 0.f;
    v = (v - mean[c]) * rsqrtf(var[c] + 1e-5f) * g[c] + be[c];
    actf[i] = v;
    actb[i] = f2bf(v);
}

// ---------------- tiny fp32 GEMM for layer 3 (N=5) ----------------
__global__ void gat_gemm_small(const float* __restrict__ A, const float* __restrict__ W,
                               float* __restrict__ C, int M, int N, int K) {
    long t = (long)blockIdx.x * blockDim.x + threadIdx.x;
    if (t >= (long)M * N) return;
    int m = (int)(t / N), j = (int)(t % N);
    const float* ap = A + (size_t)m * K;
    float s = 0.f;
    for (int k = 0; k < K; ++k) s += ap[k] * W[(size_t)k * N + j];
    C[t] = s;
}

__global__ void gat_bias_logsoftmax(const float* __restrict__ agg, const float* __restrict__ bias,
                                    float* __restrict__ out, int n) {
    int i = blockIdx.x * blockDim.x + threadIdx.x;
    if (i >= n) return;
    float v[OUTC], mx = -1e30f;
#pragma unroll
    for (int j = 0; j < OUTC; ++j) { v[j] = agg[(size_t)i * OUTC + j] + bias[j]; mx = fmaxf(mx, v[j]); }
    float s = 0.f;
#pragma unroll
    for (int j = 0; j < OUTC; ++j) s += __expf(v[j] - mx);
    float ls = logf(s);
#pragma unroll
    for (int j = 0; j < OUTC; ++j) out[(size_t)i * OUTC + j] = v[j] - mx - ls;
}

#define GRID1(n) dim3((unsigned)(((long)(n) + 255) / 256)), dim3(256), 0, stream

extern "C" void kernel_launch(void* const* d_in, const int* in_sizes, int n_in,
                              void* d_out, int out_size, void* d_ws, size_t ws_size,
                              hipStream_t stream) {
    (void)in_sizes; (void)n_in; (void)out_size; (void)ws_size;
    const float* x   = (const float*)d_in[0];
    const int*   ei  = (const int*)  d_in[1];
    const float* W1  = (const float*)d_in[2];
    const float* as1 = (const float*)d_in[3];
    const float* ad1 = (const float*)d_in[4];
    const float* b1  = (const float*)d_in[5];
    const float* W2  = (const float*)d_in[6];
    const float* as2 = (const float*)d_in[7];
    const float* ad2 = (const float*)d_in[8];
    const float* b2  = (const float*)d_in[9];
    const float* W3  = (const float*)d_in[10];
    const float* as3 = (const float*)d_in[11];
    const float* ad3 = (const float*)d_in[12];
    const float* b3  = (const float*)d_in[13];
    const float* g1  = (const float*)d_in[14];
    const float* be1 = (const float*)d_in[15];
    const float* m1  = (const float*)d_in[16];
    const float* v1  = (const float*)d_in[17];
    const float* g2  = (const float*)d_in[18];
    const float* be2 = (const float*)d_in[19];
    const float* m2  = (const float*)d_in[20];
    const float* v2  = (const float*)d_in[21];

    char* ws = (char*)d_ws;
    size_t off = 0;
    auto carve = [&](size_t bytes) { size_t o = off; off += (bytes + 255) & ~(size_t)255; return o; };
    unsigned short* bufA  = (unsigned short*)(ws + carve((size_t)NODES * INDIM * 2)); // x/act bf16
    unsigned short* bufW  = (unsigned short*)(ws + carve((size_t)INDIM * HIDC * 2));  // Wt bf16
    float*    bufH   = (float*)   (ws + carve((size_t)NODES * HIDC * 4));  // h / act fp32
    float*    bufAgg = (float*)   (ws + carve((size_t)NODES * HIDC * 4));  // aggregation
    float*    als    = (float*)   (ws + carve((size_t)NODES * HEADS * 4));
    float*    ald    = (float*)   (ws + carve((size_t)NODES * HEADS * 4));
    unsigned* emax   = (unsigned*)(ws + carve((size_t)NODES * HEADS * 4));
    float*    den    = (float*)   (ws + carve((size_t)NODES * HEADS * 4));
    float*    h3     = (float*)   (ws + carve((size_t)NODES * OUTC * 4));
    float*    agg3   = (float*)   (ws + carve((size_t)NODES * OUTC * 4));
    float*    als3   = (float*)   (ws + carve((size_t)NODES * 4));
    float*    ald3   = (float*)   (ws + carve((size_t)NODES * 4));
    unsigned* emax3  = (unsigned*)(ws + carve((size_t)NODES * 4));
    float*    den3   = (float*)   (ws + carve((size_t)NODES * 4));

    const unsigned mblocks = (unsigned)((NODES + BM - 1) / BM);

    // ---------------- layer 1: 1536 -> 8 x 64 ----------------
    gat_f32_to_bf16_v8<<<GRID1((long)NODES * INDIM / 8)>>>(
        (const float4*)x, (uint4*)bufA, (long)NODES * INDIM / 8);
    gat_f32_to_bf16_T<<<GRID1((long)INDIM * HIDC)>>>(W1, bufW, INDIM, HIDC);
    gat_wmma_gemm_tiled<<<dim3(HIDC / BN, mblocks), dim3(256), 0, stream>>>(
        bufA, bufW, bufH, NODES, HIDC, INDIM);
    gat_node_logits<<<GRID1((long)NODES * HEADS)>>>(bufH, as1, ad1, als, ald, NODES, HEADS, HIDC / HEADS);
    gat_zero_u32<<<GRID1((long)NODES * HEADS)>>>(emax, (long)NODES * HEADS);
    gat_zero_f32<<<GRID1((long)NODES * HEADS)>>>(den,  (long)NODES * HEADS);
    gat_zero_f32<<<GRID1((long)NODES * HIDC)>>>(bufAgg, (long)NODES * HIDC);
    gat_edge_max      <<<GRID1((long)EETOT * HEADS)>>>(ei, als, ald, emax, HEADS);
    gat_edge_expsum   <<<GRID1((long)EETOT * HEADS)>>>(ei, als, ald, emax, den, HEADS);
    gat_edge_aggregate<<<GRID1((long)EETOT * HEADS)>>>(ei, als, ald, emax, den, bufH, bufAgg,
                                                       HEADS, HIDC / HEADS);
    gat_bias_relu_bn<<<GRID1((long)NODES * HIDC)>>>(bufAgg, b1, g1, be1, m1, v1,
                                                    bufH, bufA, (long)NODES * HIDC, HIDC);

    // ---------------- layer 2: 512 -> 8 x 16 ----------------
    gat_f32_to_bf16_T<<<GRID1((long)HIDC * MIDC)>>>(W2, bufW, HIDC, MIDC);
    gat_wmma_gemm_tiled<<<dim3(MIDC / BN, mblocks), dim3(256), 0, stream>>>(
        bufA, bufW, bufH, NODES, MIDC, HIDC);
    gat_node_logits<<<GRID1((long)NODES * HEADS)>>>(bufH, as2, ad2, als, ald, NODES, HEADS, MIDC / HEADS);
    gat_zero_u32<<<GRID1((long)NODES * HEADS)>>>(emax, (long)NODES * HEADS);
    gat_zero_f32<<<GRID1((long)NODES * HEADS)>>>(den,  (long)NODES * HEADS);
    gat_zero_f32<<<GRID1((long)NODES * MIDC)>>>(bufAgg, (long)NODES * MIDC);
    gat_edge_max      <<<GRID1((long)EETOT * HEADS)>>>(ei, als, ald, emax, HEADS);
    gat_edge_expsum   <<<GRID1((long)EETOT * HEADS)>>>(ei, als, ald, emax, den, HEADS);
    gat_edge_aggregate<<<GRID1((long)EETOT * HEADS)>>>(ei, als, ald, emax, den, bufH, bufAgg,
                                                       HEADS, MIDC / HEADS);
    gat_bias_relu_bn<<<GRID1((long)NODES * MIDC)>>>(bufAgg, b2, g2, be2, m2, v2,
                                                    bufH, bufA, (long)NODES * MIDC, MIDC);

    // ---------------- layer 3: 128 -> 1 x 5 + log_softmax ----------------
    gat_gemm_small<<<GRID1((long)NODES * OUTC)>>>(bufH, W3, h3, NODES, OUTC, MIDC);
    gat_node_logits<<<GRID1((long)NODES)>>>(h3, as3, ad3, als3, ald3, NODES, 1, OUTC);
    gat_zero_u32<<<GRID1((long)NODES)>>>(emax3, (long)NODES);
    gat_zero_f32<<<GRID1((long)NODES)>>>(den3,  (long)NODES);
    gat_zero_f32<<<GRID1((long)NODES * OUTC)>>>(agg3, (long)NODES * OUTC);
    gat_edge_max      <<<GRID1((long)EETOT)>>>(ei, als3, ald3, emax3, 1);
    gat_edge_expsum   <<<GRID1((long)EETOT)>>>(ei, als3, ald3, emax3, den3, 1);
    gat_edge_aggregate<<<GRID1((long)EETOT)>>>(ei, als3, ald3, emax3, den3, h3, agg3, 1, OUTC);
    gat_bias_logsoftmax<<<GRID1((long)NODES)>>>(agg3, b3, (float*)d_out, NODES);
}
